// SolitonAttention_34797825032422
// MI455X (gfx1250) — matrix operand order repaired and checked by reading, not compiled, not run on hardware
//
#include <hip/hip_runtime.h>

// ---------------- types / WMMA helpers ----------------
typedef __attribute__((ext_vector_type(16))) _Float16 v16h;
typedef __attribute__((ext_vector_type(8)))  _Float16 v8h;
typedef __attribute__((ext_vector_type(8)))  float    v8f;
typedef __attribute__((ext_vector_type(4)))  unsigned v4u;
typedef __attribute__((ext_vector_type(8)))  unsigned v8u;

union AFrag { v16h v; v8h h[2]; };

static __device__ __forceinline__ v8f wmma_f16(v16h a, v16h b, v8f c) {
  // D = A(16x32 f16) x B(32x16 f16) + C(16x16 f32)
  return __builtin_amdgcn_wmma_f32_16x16x32_f16(false, a, false, b, (short)0, c, false, false);
}

// ---------------- Tensor Data Mover: 2-D tile (f16 elements) global -> LDS ----------------
// Builds a D# (groups 0+1, <=2D form) in SGPRs and issues TENSOR_LOAD_TO_LDS.
// All arguments must be wave-uniform. Caller waits with s_wait_tensorcnt.
static __device__ __forceinline__ void tdm_load_tile_f16(unsigned lds_off, const void* gaddr,
                                                         unsigned tensor_d0, unsigned tensor_d1,
                                                         unsigned tile_d0, unsigned tile_d1,
                                                         unsigned stride_d0 /* elements */) {
  unsigned long long ga = (unsigned long long)gaddr;
  v4u g0;
  g0[0] = 1u;                                                  // count=1, user D#, no gather
  g0[1] = lds_off;                                             // lds_addr (bytes)
  g0[2] = (unsigned)ga;                                        // global_addr[31:0]
  g0[3] = (unsigned)((ga >> 32) & 0x01FFFFFFu) | (2u << 30);   // global_addr[56:32] | type=2
  v8u g1;
  g1[0] = 0x00010000u;                                         // data_size=2B; no multicast/pad/iterate
  g1[1] = (tensor_d0 & 0xFFFFu) << 16;                         // tensor_dim0[15:0]
  g1[2] = (tensor_d0 >> 16) | ((tensor_d1 & 0xFFFFu) << 16);   // tensor_dim0[31:16] | tensor_dim1[15:0]
  g1[3] = (tensor_d1 >> 16) | (tile_d0 << 16);                 // tensor_dim1[31:16] | tile_dim0
  g1[4] = tile_d1;                                             // tile_dim1 | tile_dim2=0
  g1[5] = stride_d0;                                           // tensor_dim0_stride[31:0]
  g1[6] = 0u;
  g1[7] = 0u;
  asm volatile("tensor_load_to_lds %0, %1" :: "s"(g0), "s"(g1) : "memory");
}

#define BATCH 2
#define SEQ   2048
#define DM    1024
#define NH    16
#define HD    64
#define N3    3072

// ---------------- fp32 -> f16 convert ----------------
__global__ void cvt_f32_f16_kernel(const float* __restrict__ a, _Float16* __restrict__ o, int n) {
  int i = blockIdx.x * 256 + threadIdx.x;
  if (i < n) o[i] = (_Float16)a[i];
}

// ---------------- QKV GEMM: X(4096x1024) @ Wqkv(1024x3072) ----------------
// tile: M=128 (8 waves x 16), N=64 (4 subtiles of 16), K-step=32.
// A-tile DMA'd by the Tensor Data Mover, double-buffered against WMMA compute;
// B-tile transposed manually into the ping-pong buffer. One barrier/iteration.
__global__ __launch_bounds__(256) void gemm_qkv_kernel(const _Float16* __restrict__ xh,
                                                       const _Float16* __restrict__ wh,
                                                       _Float16* __restrict__ qh,
                                                       _Float16* __restrict__ kh,
                                                       float* __restrict__ vf) {
  __shared__ _Float16 As[2][128 * 32];   // [m][k]   (2 x 8 KB, written by TDM)
  __shared__ _Float16 Bs[2][64 * 32];    // transposed: [n][k]
  const int tid = threadIdx.x;
  const int wv = tid >> 5, lane = tid & 31;
  const int lrow = lane & 15, lhi = lane >> 4;
  const int Mbase = blockIdx.x * 128;
  const int Nbase = blockIdx.y * 64;

  v8f acc[4] = {};
  const int bk = tid >> 3, bn = (tid & 7) * 8;
  const unsigned lds_a0 = (unsigned)(unsigned long long)(void*)&As[0][0];
  const unsigned lds_a1 = (unsigned)(unsigned long long)(void*)&As[1][0];

  auto stageB = [&](int kt, int buf) {
    v8h bvv = *(const v8h*)&wh[(size_t)(kt * 32 + bk) * N3 + Nbase + bn];
#pragma unroll
    for (int i = 0; i < 8; ++i) Bs[buf][(bn + i) * 32 + bk] = ((_Float16*)&bvv)[i];
  };

  // prologue: stage tile 0
  if (wv == 0)
    tdm_load_tile_f16(lds_a0, &xh[(size_t)Mbase * DM], DM, BATCH * SEQ, 32, 128, DM);
  stageB(0, 0);
  if (wv == 0) __builtin_amdgcn_s_wait_tensorcnt(0);
  __syncthreads();

  for (int kt = 0; kt < 32; ++kt) {
    const int cur = kt & 1, nxt = cur ^ 1;
    if (kt + 1 < 32) {
      // overlap next-tile DMA + B transpose with this tile's WMMAs
      if (wv == 0)
        tdm_load_tile_f16(nxt ? lds_a1 : lds_a0, &xh[(size_t)Mbase * DM + (kt + 1) * 32],
                          DM, BATCH * SEQ, 32, 128, DM);
      stageB(kt + 1, nxt);
      if (kt + 2 < 32)
        __builtin_prefetch(&wh[(size_t)((kt + 2) * 32 + bk) * N3 + Nbase + bn], 0, 3);
    }
    AFrag a;
    a.h[0] = *(const v8h*)&As[cur][(wv * 16 + lrow) * 32 + lhi * 16];
    a.h[1] = *(const v8h*)&As[cur][(wv * 16 + lrow) * 32 + lhi * 16 + 8];
#pragma unroll
    for (int sub = 0; sub < 4; ++sub) {
      v16h b = *(const v16h*)&Bs[cur][(sub * 16 + lrow) * 32 + lhi * 16];
      acc[sub] = wmma_f16(a.v, b, acc[sub]);
    }
    if (wv == 0 && kt + 1 < 32) __builtin_amdgcn_s_wait_tensorcnt(0);
    __syncthreads();
  }

  // epilogue: scatter into (B,H,T,hd); whole 64-col tile shares (s,h)
  const int s = Nbase / DM;
  const int h = (Nbase % DM) / HD;
#pragma unroll
  for (int sub = 0; sub < 4; ++sub) {
    const int d = sub * 16 + lrow;
#pragma unroll
    for (int j = 0; j < 8; ++j) {
      int r = Mbase + wv * 16 + lhi * 8 + j;
      int b_ = r >> 11, t = r & 2047;
      size_t o = ((size_t)(b_ * NH + h) * SEQ + t) * HD + d;
      float val = acc[sub][j];
      if (s == 0)      qh[o] = (_Float16)val;
      else if (s == 1) kh[o] = (_Float16)val;
      else             vf[o] = val;
    }
  }
}

// ---------------- output GEMM: attn(4096x1024) @ Wout(1024x1024) -> f32 ----------------
__global__ __launch_bounds__(256) void gemm_out_kernel(const _Float16* __restrict__ ah,
                                                       const _Float16* __restrict__ wh,
                                                       float* __restrict__ out) {
  __shared__ _Float16 As[2][128 * 32];
  __shared__ _Float16 Bs[2][64 * 32];
  const int tid = threadIdx.x;
  const int wv = tid >> 5, lane = tid & 31;
  const int lrow = lane & 15, lhi = lane >> 4;
  const int Mbase = blockIdx.x * 128;
  const int Nbase = blockIdx.y * 64;

  v8f acc[4] = {};
  const int bk = tid >> 3, bn = (tid & 7) * 8;
  const unsigned lds_a0 = (unsigned)(unsigned long long)(void*)&As[0][0];
  const unsigned lds_a1 = (unsigned)(unsigned long long)(void*)&As[1][0];

  auto stageB = [&](int kt, int buf) {
    v8h bvv = *(const v8h*)&wh[(size_t)(kt * 32 + bk) * DM + Nbase + bn];
#pragma unroll
    for (int i = 0; i < 8; ++i) Bs[buf][(bn + i) * 32 + bk] = ((_Float16*)&bvv)[i];
  };

  if (wv == 0)
    tdm_load_tile_f16(lds_a0, &ah[(size_t)Mbase * DM], DM, BATCH * SEQ, 32, 128, DM);
  stageB(0, 0);
  if (wv == 0) __builtin_amdgcn_s_wait_tensorcnt(0);
  __syncthreads();

  for (int kt = 0; kt < 32; ++kt) {
    const int cur = kt & 1, nxt = cur ^ 1;
    if (kt + 1 < 32) {
      if (wv == 0)
        tdm_load_tile_f16(nxt ? lds_a1 : lds_a0, &ah[(size_t)Mbase * DM + (kt + 1) * 32],
                          DM, BATCH * SEQ, 32, 128, DM);
      stageB(kt + 1, nxt);
      if (kt + 2 < 32)
        __builtin_prefetch(&wh[(size_t)((kt + 2) * 32 + bk) * DM + Nbase + bn], 0, 3);
    }
    AFrag a;
    a.h[0] = *(const v8h*)&As[cur][(wv * 16 + lrow) * 32 + lhi * 16];
    a.h[1] = *(const v8h*)&As[cur][(wv * 16 + lrow) * 32 + lhi * 16 + 8];
#pragma unroll
    for (int sub = 0; sub < 4; ++sub) {
      v16h b = *(const v16h*)&Bs[cur][(sub * 16 + lrow) * 32 + lhi * 16];
      acc[sub] = wmma_f16(a.v, b, acc[sub]);
    }
    if (wv == 0 && kt + 1 < 32) __builtin_amdgcn_s_wait_tensorcnt(0);
    __syncthreads();
  }
#pragma unroll
  for (int sub = 0; sub < 4; ++sub)
#pragma unroll
    for (int j = 0; j < 8; ++j) {
      int r = Mbase + wv * 16 + lhi * 8 + j;
      out[(size_t)r * DM + Nbase + sub * 16 + lrow] = acc[sub][j];
    }
}

// ---------------- amax over hd per (b,h,t) ----------------
__global__ void amax_kernel(const float* __restrict__ vf, float* __restrict__ am) {
  int i = blockIdx.x * 256 + threadIdx.x;   // 0 .. B*H*T-1
  const float* p = vf + (size_t)i * HD;
  float m = 0.f;
#pragma unroll 8
  for (int d = 0; d < HD; ++d) m = fmaxf(m, fabsf(p[d]));
  am[i] = m;
}

// ---------------- soliton gate MLP per (b,h) ----------------
__global__ __launch_bounds__(64) void gate_kernel(const float* __restrict__ vf,
                                                  const float* __restrict__ gw1, const float* __restrict__ gb1,
                                                  const float* __restrict__ gw2, const float* __restrict__ gb2,
                                                  float* __restrict__ gateb) {
  __shared__ float gm[HD];
  __shared__ float hid[32];
  const int bh = blockIdx.x, d = threadIdx.x;
  const float* p = vf + (size_t)bh * SEQ * HD + d;
  float s = 0.f;
  for (int t = 0; t < SEQ; ++t) s += p[(size_t)t * HD];
  gm[d] = s * (1.0f / SEQ);
  __syncthreads();
  if (d < 32) {
    float a = gb1[d];
    for (int k = 0; k < HD; ++k) a += gm[k] * gw1[k * 32 + d];
    hid[d] = a / (1.f + __expf(-a));   // silu
  }
  __syncthreads();
  if (d == 0) {
    float o = gb2[0];
    for (int k = 0; k < 32; ++k) o += hid[k] * gw2[k];
    gateb[bh] = 1.f / (1.f + __expf(-o));
  }
}

// ---------------- PDE evolution + blend: one block per (b,h,d) channel ----------------
__global__ __launch_bounds__(256) void pde_kernel(const float* __restrict__ vf,
                                                  const float* __restrict__ amaxb,
                                                  const float* __restrict__ gateb,
                                                  const float* __restrict__ solver_mix,
                                                  _Float16* __restrict__ vbh) {
  __shared__ float smem[5 * SEQ];
  float* U  = smem;
  float* W  = smem + SEQ;
  float* TU = smem + 2 * SEQ;
  float* TW = smem + 3 * SEQ;
  float* G  = smem + 4 * SEQ;
  const int tid = threadIdx.x;
  const int ch = blockIdx.x;
  const int bh = ch >> 6;
  const int d  = ch & 63;

  float s0 = solver_mix[0], s1 = solver_mix[1], s2 = solver_mix[2];
  float sm = fmaxf(s0, fmaxf(s1, s2));
  float e0 = __expf(s0 - sm), e1 = __expf(s1 - sm), e2 = __expf(s2 - sm);
  float inv = 1.f / (e0 + e1 + e2);
  float mw0 = e0 * inv, mw1 = e1 * inv, mw2 = e2 * inv;

  float u0r[8], vraw[8], accm[8];
#pragma unroll
  for (int i = 0; i < 8; ++i) {
    int t = tid + i * 256;
    float a = amaxb[bh * SEQ + t];
    a = fmaxf(a, 1e-6f) + 1e-6f;
    float v = vf[((size_t)bh * SEQ + t) * HD + d];
    vraw[i] = v;
    u0r[i]  = v * 2.0f / a;
    accm[i] = 0.f;
  }

  // f evaluators (all threads execute uniformly; HJ has internal barriers)
  auto sg_f = [&](const float* A, const float* Wp, float* fu, float* fw) {
#pragma unroll
    for (int i = 0; i < 8; ++i) {
      int t = tid + i * 256, tp = (t + 1) & 2047, tm = (t - 1) & 2047;
      float u = A[t], w = Wp[t];
      fu[i] = w;
      fw[i] = A[tp] - 2.f * u + A[tm] - __sinf(u) - 0.01f * w;
    }
  };
  auto hj_f = [&](const float* A, const float* Wp, float* fu, float* fw) {
    __syncthreads();
#pragma unroll
    for (int i = 0; i < 8; ++i) {
      int t = tid + i * 256, tp = (t + 1) & 2047, tm = (t - 1) & 2047;
      float u = A[t];
      G[t] = (1.f - u + u * u) * (A[tp] - A[tm]) * 0.5f;
    }
    __syncthreads();
#pragma unroll
    for (int i = 0; i < 8; ++i) {
      int t = tid + i * 256, tp = (t + 1) & 2047, tm = (t - 1) & 2047;
      int tp2 = (t + 2) & 2047, tm2 = (t - 2) & 2047;
      fu[i] = Wp[t];
      fw[i] = (G[tp] - G[tm]) * 0.5f
              - 0.1f * (A[tp2] - 4.f * A[tp] + 6.f * A[t] - 4.f * A[tm] + A[tm2])
              - 0.01f * Wp[t];
    }
  };
  auto kdv_f = [&](const float* A, float* fu) {
#pragma unroll
    for (int i = 0; i < 8; ++i) {
      int t = tid + i * 256, tp = (t + 1) & 2047, tm = (t - 1) & 2047;
      int tp2 = (t + 2) & 2047, tm2 = (t - 2) & 2047;
      fu[i] = -(6.f * A[t] * (A[tp] - A[tm]) * 0.5f
                + (A[tp2] - 2.f * A[tp] + 2.f * A[tm] - A[tm2]) * 0.5f)
              - 0.01f * A[t];
    }
  };

  auto rk4_2f = [&](auto&& f, float dtv) {
    for (int step = 0; step < 3; ++step) {
      float au[8], aw[8], fu[8], fw[8];
      f(U, W, fu, fw);
#pragma unroll
      for (int i = 0; i < 8; ++i) { au[i] = fu[i]; aw[i] = fw[i]; }
      __syncthreads();
#pragma unroll
      for (int i = 0; i < 8; ++i) { int t = tid + i * 256; TU[t] = U[t] + 0.5f * dtv * fu[i]; TW[t] = W[t] + 0.5f * dtv * fw[i]; }
      __syncthreads();
      f(TU, TW, fu, fw);
#pragma unroll
      for (int i = 0; i < 8; ++i) { au[i] += 2.f * fu[i]; aw[i] += 2.f * fw[i]; }
      __syncthreads();
#pragma unroll
      for (int i = 0; i < 8; ++i) { int t = tid + i * 256; TU[t] = U[t] + 0.5f * dtv * fu[i]; TW[t] = W[t] + 0.5f * dtv * fw[i]; }
      __syncthreads();
      f(TU, TW, fu, fw);
#pragma unroll
      for (int i = 0; i < 8; ++i) { au[i] += 2.f * fu[i]; aw[i] += 2.f * fw[i]; }
      __syncthreads();
#pragma unroll
      for (int i = 0; i < 8; ++i) { int t = tid + i * 256; TU[t] = U[t] + dtv * fu[i]; TW[t] = W[t] + dtv * fw[i]; }
      __syncthreads();
      f(TU, TW, fu, fw);
#pragma unroll
      for (int i = 0; i < 8; ++i) { au[i] += fu[i]; aw[i] += fw[i]; }
      __syncthreads();
#pragma unroll
      for (int i = 0; i < 8; ++i) { int t = tid + i * 256; U[t] += (dtv / 6.f) * au[i]; W[t] += (dtv / 6.f) * aw[i]; }
      __syncthreads();
    }
  };
  auto rk4_1f = [&](auto&& f, float dtv) {
    for (int step = 0; step < 3; ++step) {
      float au[8], fu[8];
      f(U, fu);
#pragma unroll
      for (int i = 0; i < 8; ++i) au[i] = fu[i];
      __syncthreads();
#pragma unroll
      for (int i = 0; i < 8; ++i) { int t = tid + i * 256; TU[t] = U[t] + 0.5f * dtv * fu[i]; }
      __syncthreads();
      f(TU, fu);
#pragma unroll
      for (int i = 0; i < 8; ++i) au[i] += 2.f * fu[i];
      __syncthreads();
#pragma unroll
      for (int i = 0; i < 8; ++i) { int t = tid + i * 256; TU[t] = U[t] + 0.5f * dtv * fu[i]; }
      __syncthreads();
      f(TU, fu);
#pragma unroll
      for (int i = 0; i < 8; ++i) au[i] += 2.f * fu[i];
      __syncthreads();
#pragma unroll
      for (int i = 0; i < 8; ++i) { int t = tid + i * 256; TU[t] = U[t] + dtv * fu[i]; }
      __syncthreads();
      f(TU, fu);
#pragma unroll
      for (int i = 0; i < 8; ++i) au[i] += fu[i];
      __syncthreads();
#pragma unroll
      for (int i = 0; i < 8; ++i) { int t = tid + i * 256; U[t] += (dtv / 6.f) * au[i]; }
      __syncthreads();
    }
  };

  // sine-Gordon (dt = 0.05)
#pragma unroll
  for (int i = 0; i < 8; ++i) { int t = tid + i * 256; U[t] = u0r[i]; W[t] = 0.f; }
  __syncthreads();
  rk4_2f(sg_f, 0.05f);
#pragma unroll
  for (int i = 0; i < 8; ++i) accm[i] += mw0 * U[tid + i * 256];

  // KdV (dt = 0.025)
  __syncthreads();
#pragma unroll
  for (int i = 0; i < 8; ++i) { int t = tid + i * 256; U[t] = u0r[i]; }
  __syncthreads();
  rk4_1f(kdv_f, 0.025f);
#pragma unroll
  for (int i = 0; i < 8; ++i) accm[i] += mw1 * U[tid + i * 256];

  // Heimburg-Jackson (dt = 0.025)
  __syncthreads();
#pragma unroll
  for (int i = 0; i < 8; ++i) { int t = tid + i * 256; U[t] = u0r[i]; W[t] = 0.f; }
  __syncthreads();
  rk4_2f(hj_f, 0.025f);
#pragma unroll
  for (int i = 0; i < 8; ++i) accm[i] += mw2 * U[tid + i * 256];

  // blend with gate, write f16 V
  float g = gateb[bh];
#pragma unroll
  for (int i = 0; i < 8; ++i) {
    int t = tid + i * 256;
    vbh[((size_t)bh * SEQ + t) * HD + d] = (_Float16)(g * accm[i] + (1.f - g) * vraw[i]);
  }
}

// ---------------- flash causal attention: block = 128 q rows, 8 waves x 16 rows ----------------
// K tile DMA'd by TDM (double-buffered); V tile transposed manually (double-buffered).
__global__ __launch_bounds__(256) void attn_kernel(const _Float16* __restrict__ qh,
                                                   const _Float16* __restrict__ kh,
                                                   const _Float16* __restrict__ vbh,
                                                   _Float16* __restrict__ attnh) {
  __shared__ _Float16 ldsK[2][32 * 64];    // [key][hd]   (TDM-staged)
  __shared__ _Float16 ldsVt[2][64 * 32];   // [hd][key]
  __shared__ _Float16 ldsP[8][16 * 32];    // per-wave P staging
  const int tid = threadIdx.x, wv = tid >> 5, lane = tid & 31;
  const int lrow = lane & 15, lhi = lane >> 4;
  const int bh = blockIdx.y, b = bh >> 4, h = bh & 15;
  const int qbase = blockIdx.x * 128;
  const int qw = qbase + wv * 16;
  const float scale = 0.125f;  // 1/sqrt(64)
  const unsigned lds_k0 = (unsigned)(unsigned long long)(void*)&ldsK[0][0];
  const unsigned lds_k1 = (unsigned)(unsigned long long)(void*)&ldsK[1][0];

  // Q fragments (A-matrix layout: contiguous 16-half chunk per lane half)
  AFrag aq[2];
  const _Float16* qp = qh + ((size_t)bh * SEQ + qw + lrow) * HD;
#pragma unroll
  for (int kc = 0; kc < 2; ++kc) {
    aq[kc].h[0] = *(const v8h*)(qp + kc * 32 + lhi * 16);
    aq[kc].h[1] = *(const v8h*)(qp + kc * 32 + lhi * 16 + 8);
  }

  v8f ov[4] = {};
  float mrow[8], lsum[8];
#pragma unroll
  for (int j = 0; j < 8; ++j) { mrow[j] = -1e30f; lsum[j] = 0.f; }

  const int kend = qbase + 128;
  const int skey = tid >> 3, soff = (tid & 7) * 8;

  auto stageV = [&](int kb, int buf) {
    v8h vv = *(const v8h*)&vbh[((size_t)bh * SEQ + kb + skey) * HD + soff];
#pragma unroll
    for (int i = 0; i < 8; ++i) ldsVt[buf][(soff + i) * 32 + skey] = ((_Float16*)&vv)[i];
  };

  // prologue: stage key block 0
  if (wv == 0)
    tdm_load_tile_f16(lds_k0, &kh[(size_t)bh * SEQ * HD], HD, SEQ, HD, 32, HD);
  stageV(0, 0);
  if (wv == 0) __builtin_amdgcn_s_wait_tensorcnt(0);
  __syncthreads();

  for (int kb = 0; kb < kend; kb += 32) {
    const int cur = (kb >> 5) & 1, nxt = cur ^ 1;
    if (kb + 32 < kend) {
      // overlap next key-block staging (TDM + V transpose) with this block's WMMAs
      if (wv == 0)
        tdm_load_tile_f16(nxt ? lds_k1 : lds_k0, &kh[((size_t)bh * SEQ + kb + 32) * HD],
                          HD, SEQ, HD, 32, HD);
      stageV(kb + 32, nxt);
      if (kb + 64 < kend)
        __builtin_prefetch(&vbh[((size_t)bh * SEQ + kb + 64 + skey) * HD + soff], 0, 3);
    }

    if (kb <= qw + 15) {                 // skip fully-masked key blocks (barriers stay uniform)
      v8f sA = {}, sB = {};
#pragma unroll
      for (int kc = 0; kc < 2; ++kc) {
        v16h b0 = *(const v16h*)&ldsK[cur][lrow * 64 + kc * 32 + lhi * 16];
        v16h b1 = *(const v16h*)&ldsK[cur][(16 + lrow) * 64 + kc * 32 + lhi * 16];
        sA = wmma_f16(aq[kc].v, b0, sA);
        sB = wmma_f16(aq[kc].v, b1, sB);
      }
      float p0[8], p1[8], corr[8];
#pragma unroll
      for (int j = 0; j < 8; ++j) {
        const int qrow = qw + lhi * 8 + j;
        float x0 = sA[j] * scale, x1 = sB[j] * scale;
        if (kb + lrow > qrow)      x0 = -1e30f;   // causal mask
        if (kb + 16 + lrow > qrow) x1 = -1e30f;
        p0[j] = x0; p1[j] = x1;
      }
#pragma unroll
      for (int j = 0; j < 8; ++j) {
        float r = fmaxf(p0[j], p1[j]);
        r = fmaxf(r, __shfl_xor(r, 1, 32));
        r = fmaxf(r, __shfl_xor(r, 2, 32));
        r = fmaxf(r, __shfl_xor(r, 4, 32));
        r = fmaxf(r, __shfl_xor(r, 8, 32));
        float mnew = fmaxf(mrow[j], r);
        corr[j] = __expf(mrow[j] - mnew);
        mrow[j] = mnew;
        p0[j] = __expf(p0[j] - mnew);
        p1[j] = __expf(p1[j] - mnew);
        float rs = p0[j] + p1[j];
        rs += __shfl_xor(rs, 1, 32);
        rs += __shfl_xor(rs, 2, 32);
        rs += __shfl_xor(rs, 4, 32);
        rs += __shfl_xor(rs, 8, 32);
        lsum[j] = lsum[j] * corr[j] + rs;
      }
      // relayout P (C-layout -> A-layout) through per-wave LDS
      _Float16* pw = &ldsP[wv][0];
#pragma unroll
      for (int j = 0; j < 8; ++j) {
        pw[(lhi * 8 + j) * 32 + lrow]      = (_Float16)p0[j];
        pw[(lhi * 8 + j) * 32 + 16 + lrow] = (_Float16)p1[j];
      }
#pragma unroll
      for (int sub = 0; sub < 4; ++sub)
#pragma unroll
        for (int j = 0; j < 8; ++j) ov[sub][j] *= corr[j];
      asm volatile("s_wait_dscnt 0" ::: "memory");   // wave-local LDS RAW
      AFrag ap;
      ap.h[0] = *(const v8h*)&pw[lrow * 32 + lhi * 16];
      ap.h[1] = *(const v8h*)&pw[lrow * 32 + lhi * 16 + 8];
#pragma unroll
      for (int sub = 0; sub < 4; ++sub) {
        v16h bv = *(const v16h*)&ldsVt[cur][(sub * 16 + lrow) * 32 + lhi * 16];
        ov[sub] = wmma_f16(ap.v, bv, ov[sub]);
      }
    }

    if (wv == 0 && kb + 32 < kend) __builtin_amdgcn_s_wait_tensorcnt(0);
    __syncthreads();
  }

  // normalize and write f16 (B,T,D) for the output GEMM
#pragma unroll
  for (int j = 0; j < 8; ++j) {
    float inv = 1.0f / lsum[j];
    int t = qw + lhi * 8 + j;
    size_t o = ((size_t)b * SEQ + t) * DM + h * HD;
#pragma unroll
    for (int sub = 0; sub < 4; ++sub)
      attnh[o + sub * 16 + lrow] = (_Float16)(ov[sub][j] * inv);
  }
}

// ---------------- host-side launch ----------------
extern "C" void kernel_launch(void* const* d_in, const int* in_sizes, int n_in,
                              void* d_out, int out_size, void* d_ws, size_t ws_size,
                              hipStream_t stream) {
  (void)in_sizes; (void)n_in; (void)out_size; (void)ws_size;
  const float* x          = (const float*)d_in[0];
  const float* w_qkv      = (const float*)d_in[1];
  const float* w_out      = (const float*)d_in[2];
  const float* solver_mix = (const float*)d_in[3];
  const float* gw1        = (const float*)d_in[4];
  const float* gb1        = (const float*)d_in[5];
  const float* gw2        = (const float*)d_in[6];
  const float* gb2        = (const float*)d_in[7];
  float* out = (float*)d_out;

  char* ws = (char*)d_ws;
  constexpr size_t MB = 1024 * 1024;
  _Float16* xh    = (_Float16*)(ws + 0);         //  8 MB  (4096x1024 f16)
  _Float16* wqkvh = (_Float16*)(ws + 8  * MB);   //  6 MB  (1024x3072 f16)
  _Float16* wouth = (_Float16*)(ws + 14 * MB);   //  2 MB  (1024x1024 f16)
  _Float16* qh    = (_Float16*)(ws + 16 * MB);   //  8 MB  (B,H,T,hd f16)
  _Float16* kh    = (_Float16*)(ws + 24 * MB);   //  8 MB
  float*    vf    = (float*)   (ws + 32 * MB);   // 16 MB  (B,H,T,hd f32)
  _Float16* vbh   = (_Float16*)(ws + 48 * MB);   //  8 MB  (blended V f16)
  _Float16* attnh = (_Float16*)(ws + 56 * MB);   //  8 MB  (B,T,D f16)
  float*    amaxb = (float*)   (ws + 64 * MB);   // 256 KB (B,H,T)
  float*    gateb = (float*)   (ws + 64 * MB + 256 * 1024); // 128 B

  const int nX = BATCH * SEQ * DM;      // 4194304
  const int nWq = DM * N3;              // 3145728
  const int nWo = DM * DM;              // 1048576
  cvt_f32_f16_kernel<<<(nX + 255) / 256, 256, 0, stream>>>(x, xh, nX);
  cvt_f32_f16_kernel<<<(nWq + 255) / 256, 256, 0, stream>>>(w_qkv, wqkvh, nWq);
  cvt_f32_f16_kernel<<<(nWo + 255) / 256, 256, 0, stream>>>(w_out, wouth, nWo);

  gemm_qkv_kernel<<<dim3(32, 48), 256, 0, stream>>>(xh, wqkvh, qh, kh, vf);

  amax_kernel<<<(BATCH * NH * SEQ) / 256, 256, 0, stream>>>(vf, amaxb);
  gate_kernel<<<BATCH * NH, 64, 0, stream>>>(vf, gw1, gb1, gw2, gb2, gateb);
  pde_kernel<<<BATCH * NH * HD, 256, 0, stream>>>(vf, amaxb, gateb, solver_mix, vbh);

  attn_kernel<<<dim3(SEQ / 128, BATCH * NH), 256, 0, stream>>>(qh, kh, vbh, attnh);

  gemm_out_kernel<<<dim3(32, 16), 256, 0, stream>>>(attnh, wouth, out);
}
